// PIGNN_77464030151232
// MI455X (gfx1250) — matrix operand order repaired and checked by reading, not compiled, not run on hardware
//
#include <hip/hip_runtime.h>
#include <hip/hip_bf16.h>
#include <stdint.h>

#define N_NODES 10000
#define N_EDGES 80000
#define NODE_IN 25
#define EDGE_IN 3
#define H 128
#define STEPS 12
#define MSG_IN 259   // 2*H + EDGE_IN
#define MSG_K 288    // padded to 9*32
#define UPD_IN 256
#define EMB_IN 28
#define EMB_K 32     // padded to 32
#define TW 20

typedef __attribute__((ext_vector_type(16))) __bf16 v16bf;
typedef __attribute__((ext_vector_type(8)))  float  v8f;

// Wave-local LDS ordering (wave-private LDS slices, no workgroup barrier needed).
#define LDS_FENCE()   asm volatile("s_wait_dscnt 0" ::: "memory")
// Async (memory->LDS) completion; ASYNCcnt is unordered vs DS ops.
#define ASYNC_FENCE() asm volatile("s_wait_asynccnt 0x0" ::: "memory")

// Direct memory->LDS 16-byte copy via the CDNA5 async path (no VGPR data).
// dsaddr = LDS address (low 32 bits of generic pointer), gaddr = 64-bit global.
__device__ __forceinline__ void async_copy_b128(unsigned lds_addr, const void* gptr) {
  asm volatile("global_load_async_to_lds_b128 %0, %1, off"
               :: "v"(lds_addr), "v"((unsigned long long)(uintptr_t)gptr)
               : "memory");
}

// ---------------------------------------------------------------------------
// Packed fragment layouts (CDNA5 ISA §7.12.2):
// A 16x32 bf16 tile: lanes 0-15 hold M=lane, K in {0..7}u{16..23};
//                    lanes16-31 hold M=lane-16, K in {8..15}u{24..31};
//                    VGPR v<4 -> K=2v,2v+1 ; v>=4 -> K=16+2(v-4)+..
// Packed-A: element (m,k) stored at ((ks*32)+lane)*16 + 2v+p so each lane's
// fragment is 32 contiguous bytes -> 2x ds_load_b128. Any aligned 8-element
// k-block of one row is contiguous (one half-fragment) -> async b128 staging.
__device__ __forceinline__ int a_pack_off(int m, int k) {
  int ks = k >> 5, k32 = k & 31;
  int laneHalf = (k32 >> 3) & 1;                 // K 8..15 / 24..31 -> upper lanes
  int base = k32 - (laneHalf << 3);              // {0..7} u {16..23}
  int v = (base < 8) ? (base >> 1) : (4 + ((base - 16) >> 1));
  int p = base & 1;
  int lane = m + (laneHalf << 4);
  return ((ks << 5) + lane) * 16 + (v << 1) + p;
}

// B 32x16 bf16 tile: lanes 0-15 hold K in {0..7}u{16..23} at N=lane (same v->K map as A),
// lanes 16-31 hold K in {8..15}u{24..31}. Weights pre-packed so each lane's fragment is
// 32 contiguous bytes -> 2x global_load_b128.

// 16 x 128 output tile = 8 accumulators; K = KSTEPS*32.
template <int KSTEPS>
__device__ __forceinline__ void gemm_tile(const __bf16* A, const __bf16* Wp,
                                          v8f acc[8], int lane) {
#pragma unroll
  for (int nt = 0; nt < 8; ++nt)
#pragma unroll
    for (int i = 0; i < 8; ++i) acc[nt][i] = 0.f;
#pragma unroll
  for (int ks = 0; ks < KSTEPS; ++ks) {
    v16bf a = *(const v16bf*)(A + (((ks << 5) + lane) << 4));
#pragma unroll
    for (int nt = 0; nt < 8; ++nt) {
      v16bf b = *(const v16bf*)(Wp + ((((long long)ks * 8 + nt) * 32 + lane) << 4));
      acc[nt] = __builtin_amdgcn_wmma_f32_16x16x32_bf16(
          false, a, false, b, (short)0, acc[nt], false, false);
    }
  }
}

// C/D layout: VGPR r -> (m = r + 8*(lane>=16), n = nt*16 + (lane&15)).
// Write relu(acc+bias) into LDS in *packed-A* order for the next GEMM (k = n).
__device__ __forceinline__ void store_relu_lds_packed(const v8f acc[8], const float* bias,
                                                      __bf16* out, int lane) {
  int nbase = lane & 15;
  int mh    = (lane >> 4) << 3;
#pragma unroll
  for (int nt = 0; nt < 8; ++nt) {
    int n = nt * 16 + nbase;
    float bv = bias[n];
#pragma unroll
    for (int r = 0; r < 8; ++r) {
      float v = acc[nt][r] + bv;
      v = v > 0.f ? v : 0.f;
      out[a_pack_off(mh + r, n)] = (__bf16)v;
    }
  }
}

// ---------------------------------------------------------------------------
// Utility kernels
__global__ void zero_f32_kernel(float* p, long long n) {
  long long i = (long long)blockIdx.x * blockDim.x + threadIdx.x;
  if (i < n) p[i] = 0.f;
}

// Pack one (or STEPS) KxH f32 weight matrices into bf16 B-fragment order,
// zero-padding K up to ksteps*32.
__global__ void pack_b_bf16_kernel(const float* __restrict__ src, __bf16* __restrict__ dst,
                                   int src_rows, int ksteps) {
  int z = blockIdx.y;
  long long tot = (long long)ksteps * 8 * 32 * 16;
  long long i = (long long)blockIdx.x * blockDim.x + threadIdx.x;
  if (i >= tot) return;
  int e    = (int)(i & 15);
  int lane = (int)((i >> 4) & 31);
  int nt   = (int)((i >> 9) & 7);
  int ks   = (int)(i >> 12);
  int v = e >> 1, p = e & 1;
  int n = nt * 16 + (lane & 15);
  int k = ks * 32 + ((lane >> 4) << 3) + ((v >> 2) << 4) + 2 * (v & 3) + p;
  float val = (k < src_rows) ? src[((long long)z * src_rows + k) * H + n] : 0.f;
  dst[(long long)z * tot + i] = (__bf16)val;
}

__global__ void deg_kernel(const int* __restrict__ ei, float* __restrict__ deg, int e_cnt) {
  int i = blockIdx.x * blockDim.x + threadIdx.x;
  if (i < e_cnt) atomicAdd(&deg[ei[e_cnt + i]], 1.0f);   // dst = edge_index[1]
}

__global__ void invdeg_kernel(const float* __restrict__ deg, float* __restrict__ inv, int n) {
  int i = blockIdx.x * blockDim.x + threadIdx.x;
  if (i < n) {
    float d = deg[i];
    inv[i] = d > 0.f ? 1.f / d : 0.f;
  }
}

// ---------------------------------------------------------------------------
// Node embedding: h = relu(relu([x,posx,posy,g] @ W1 + b1) @ W2 + b2)
__global__ __launch_bounds__(128) void embed_kernel(
    const float* __restrict__ x, const float* __restrict__ pos, const float* __restrict__ gattr,
    const __bf16* __restrict__ W1, const float* __restrict__ b1,
    const __bf16* __restrict__ W2, const float* __restrict__ b2,
    float* __restrict__ hf, __bf16* __restrict__ hbf, int n, int n_tiles) {
  __shared__ __attribute__((aligned(32))) __bf16 lds[4][16 * H];
  int lane = threadIdx.x & 31, wid = threadIdx.x >> 5;
  int tile = blockIdx.x * 4 + wid;
  if (tile >= n_tiles) return;                 // wave-uniform
  __bf16* A = lds[wid];

  for (int idx = lane; idx < 16 * EMB_K; idx += 32) {
    int r = idx >> 5, c = idx & 31;
    int node = tile * 16 + r; if (node >= n) node = n - 1;
    float v;
    if (c < NODE_IN)            v = x[(long long)node * NODE_IN + c];
    else if (c == NODE_IN)      v = pos[(long long)node * 2 + 0];
    else if (c == NODE_IN + 1)  v = pos[(long long)node * 2 + 1];
    else if (c == NODE_IN + 2)  v = gattr[0];
    else                        v = 0.f;
    A[a_pack_off(r, c)] = (__bf16)v;
  }
  LDS_FENCE();

  v8f acc[8];
  gemm_tile<1>(A, W1, acc, lane);
  store_relu_lds_packed(acc, b1, A, lane);
  LDS_FENCE();
  gemm_tile<4>(A, W2, acc, lane);

  int nbase = lane & 15, mh = (lane >> 4) << 3;
#pragma unroll
  for (int nt = 0; nt < 8; ++nt) {
    int nn = nt * 16 + nbase;
    float bv = b2[nn];
#pragma unroll
    for (int r = 0; r < 8; ++r) {
      int node = tile * 16 + mh + r;
      if (node < n) {
        float v = acc[nt][r] + bv; v = v > 0.f ? v : 0.f;
        hf[(long long)node * H + nn]  = v;
        hbf[(long long)node * H + nn] = (__bf16)v;
      }
    }
  }
}

// ---------------------------------------------------------------------------
// Edge messages: m = relu(relu([h_dst,h_src,ea] @ W1 + b1) @ W2 + b2); agg[dst]+=m
__global__ __launch_bounds__(128) void msg_kernel(
    const __bf16* __restrict__ hbf, const float* __restrict__ eattr, const int* __restrict__ ei,
    const __bf16* __restrict__ W1, const float* __restrict__ b1,
    const __bf16* __restrict__ W2, const float* __restrict__ b2,
    float* __restrict__ agg, int e_cnt, int n_tiles) {
  __shared__ __attribute__((aligned(32))) __bf16 lds[4][16 * MSG_K];
  __shared__ int dsts[4][16];
  int lane = threadIdx.x & 31, wid = threadIdx.x >> 5;
  int tile = blockIdx.x * 4 + wid;
  if (tile >= n_tiles) return;
  __bf16* A = lds[wid];

  // Tail columns 256..287 (edge_attr + zero pad) + per-row dst stash (VGPR path)
  for (int r = 0; r < 16; ++r) {
    int e = tile * 16 + r;
    bool valid = e < e_cnt;
    if (!valid) e = 0;
    if (lane == 0) dsts[wid][r] = valid ? ei[e_cnt + e] : -1;
    int c = 2 * H + lane;                       // 256..287, one per lane
    __bf16 v = (c < MSG_IN) ? (__bf16)eattr[(long long)e * EDGE_IN + (c - 2 * H)]
                            : (__bf16)0.f;
    A[a_pack_off(r, c)] = v;
  }

  // h[dst] | h[src] gather (cols 0..255): direct memory->LDS async b128 copies.
  // 16 rows x 32 16B-blocks = 512 descriptors, 16 per lane.
  for (int idx = lane; idx < 512; idx += 32) {
    int r = idx >> 5, t = idx & 31;             // t<16 -> h[dst], else h[src]
    int e = tile * 16 + r;
    if (e >= e_cnt) e = 0;
    int node = (t < 16) ? ei[e_cnt + e] : ei[e];
    const __bf16* src = hbf + (long long)node * H + ((t & 15) << 3);
    unsigned la = (unsigned)(uintptr_t)(A + a_pack_off(r, t << 3));
    async_copy_b128(la, src);
  }
  ASYNC_FENCE();
  LDS_FENCE();

  v8f acc[8];
  gemm_tile<9>(A, W1, acc, lane);              // K = 288
  store_relu_lds_packed(acc, b1, A, lane);
  LDS_FENCE();
  gemm_tile<4>(A, W2, acc, lane);              // K = 128

  int nbase = lane & 15, mh = (lane >> 4) << 3;
#pragma unroll
  for (int nt = 0; nt < 8; ++nt) {
    int nn = nt * 16 + nbase;
    float bv = b2[nn];
#pragma unroll
    for (int r = 0; r < 8; ++r) {
      int dst = dsts[wid][mh + r];
      if (dst >= 0) {
        float v = acc[nt][r] + bv; v = v > 0.f ? v : 0.f;
        atomicAdd(&agg[(long long)dst * H + nn], v);
      }
    }
  }
}

// ---------------------------------------------------------------------------
// Node update: u = relu(relu([h, agg*inv_deg] @ U1 + u1) @ U2 + u2); h += u; agg = 0
__global__ __launch_bounds__(128) void upd_kernel(
    const float* __restrict__ inv_deg, float* __restrict__ agg,
    float* __restrict__ hf, __bf16* __restrict__ hbf,
    const __bf16* __restrict__ U1, const float* __restrict__ u1b,
    const __bf16* __restrict__ U2, const float* __restrict__ u2b, int n, int n_tiles) {
  __shared__ __attribute__((aligned(32))) __bf16 lds[4][16 * UPD_IN];
  int lane = threadIdx.x & 31, wid = threadIdx.x >> 5;
  int tile = blockIdx.x * 4 + wid;
  if (tile >= n_tiles) return;
  __bf16* A = lds[wid];

  // h part (cols 0..127): async memory->LDS b128; 16 rows x 16 blocks.
  for (int idx = lane; idx < 256; idx += 32) {
    int r = idx >> 4, t = idx & 15;
    int node = tile * 16 + r; if (node >= n) node = n - 1;
    const __bf16* src = hbf + (long long)node * H + (t << 3);
    unsigned la = (unsigned)(uintptr_t)(A + a_pack_off(r, t << 3));
    async_copy_b128(la, src);
  }
  // agg part (cols 128..255): needs *inv_deg scale + zeroing -> VGPR path.
  for (int r = 0; r < 16; ++r) {
    int node = tile * 16 + r; if (node >= n) node = n - 1;
    float id = inv_deg[node];
    float* an = agg + (long long)node * H;
    for (int c = H + lane; c < UPD_IN; c += 32) {
      float a = an[c - H] * id;
      A[a_pack_off(r, c)] = (__bf16)a;
      an[c - H] = 0.f;                          // clear for next step
    }
  }
  ASYNC_FENCE();
  LDS_FENCE();

  v8f acc[8];
  gemm_tile<8>(A, U1, acc, lane);              // K = 256
  store_relu_lds_packed(acc, u1b, A, lane);
  LDS_FENCE();
  gemm_tile<4>(A, U2, acc, lane);              // K = 128

  int nbase = lane & 15, mh = (lane >> 4) << 3;
#pragma unroll
  for (int nt = 0; nt < 8; ++nt) {
    int nn = nt * 16 + nbase;
    float bv = u2b[nn];
#pragma unroll
    for (int r = 0; r < 8; ++r) {
      int node = tile * 16 + mh + r;
      if (node < n) {
        float u = acc[nt][r] + bv; u = u > 0.f ? u : 0.f;
        float nh = hf[(long long)node * H + nn] + u;
        hf[(long long)node * H + nn]  = nh;
        hbf[(long long)node * H + nn] = (__bf16)nh;
      }
    }
  }
}

// ---------------------------------------------------------------------------
// Conv head: (N,1,128) -conv1(k15,s4,8ch)-> relu -> (N,8,29) -conv2(k10)-> (N,20)
__global__ __launch_bounds__(128) void conv_kernel(
    const float* __restrict__ hf,
    const float* __restrict__ c1W, const float* __restrict__ c1b,
    const float* __restrict__ c2W, const float* __restrict__ c2b,
    float* __restrict__ out, int n) {
  __shared__ float hrow[4][H];
  __shared__ float y1[4][8 * 29];
  int lane = threadIdx.x & 31, wid = threadIdx.x >> 5;
  int node = blockIdx.x * 4 + wid;
  if (node >= n) return;

  for (int c = lane; c < H; c += 32) hrow[wid][c] = hf[(long long)node * H + c];
  LDS_FENCE();

  for (int idx = lane; idx < 8 * 29; idx += 32) {
    int c = idx / 29, p = idx % 29;
    float a = c1b[c];
#pragma unroll
    for (int j = 0; j < 15; ++j) a += c1W[c * 15 + j] * hrow[wid][p * 4 + j];
    y1[wid][idx] = a > 0.f ? a : 0.f;
  }
  LDS_FENCE();

  if (lane < TW) {
    float a = c2b[0];
#pragma unroll
    for (int c = 0; c < 8; ++c)
#pragma unroll
      for (int k = 0; k < 10; ++k)
        a += c2W[c * 10 + k] * y1[wid][c * 29 + lane + k];
    out[(long long)node * TW + lane] = a;
  }
}

// ---------------------------------------------------------------------------
extern "C" void kernel_launch(void* const* d_in, const int* in_sizes, int n_in,
                              void* d_out, int out_size, void* d_ws, size_t ws_size,
                              hipStream_t stream) {
  (void)in_sizes; (void)n_in; (void)out_size; (void)ws_size;
  const float* x       = (const float*)d_in[0];
  const float* pos     = (const float*)d_in[1];
  const float* gattr   = (const float*)d_in[2];
  const float* eattr   = (const float*)d_in[3];
  const float* emb_W1  = (const float*)d_in[4];
  const float* emb_b1  = (const float*)d_in[5];
  const float* emb_W2  = (const float*)d_in[6];
  const float* emb_b2  = (const float*)d_in[7];
  const float* msg1_W  = (const float*)d_in[8];
  const float* msg1_b  = (const float*)d_in[9];
  const float* msg2_W  = (const float*)d_in[10];
  const float* msg2_b  = (const float*)d_in[11];
  const float* upd1_W  = (const float*)d_in[12];
  const float* upd1_b  = (const float*)d_in[13];
  const float* upd2_W  = (const float*)d_in[14];
  const float* upd2_b  = (const float*)d_in[15];
  const float* c1W     = (const float*)d_in[16];
  const float* c1b     = (const float*)d_in[17];
  const float* c2W     = (const float*)d_in[18];
  const float* c2b     = (const float*)d_in[19];
  const int*   ei      = (const int*)d_in[20];
  float* out = (float*)d_out;

  char* ws = (char*)d_ws;
  size_t off = 0;
  auto walloc = [&](size_t bytes) -> void* {
    void* p = (void*)(ws + off);
    off += (bytes + 255) & ~(size_t)255;
    return p;
  };
  float*  hf   = (float*)walloc((size_t)N_NODES * H * 4);
  float*  agg  = (float*)walloc((size_t)N_NODES * H * 4);
  float*  deg  = (float*)walloc((size_t)N_NODES * 4);
  float*  invd = (float*)walloc((size_t)N_NODES * 4);
  __bf16* hbf  = (__bf16*)walloc((size_t)N_NODES * H * 2);
  __bf16* wE1  = (__bf16*)walloc((size_t)EMB_K * H * 2);
  __bf16* wE2  = (__bf16*)walloc((size_t)H * H * 2);
  __bf16* wM1  = (__bf16*)walloc((size_t)STEPS * MSG_K * H * 2);
  __bf16* wM2  = (__bf16*)walloc((size_t)STEPS * H * H * 2);
  __bf16* wU1  = (__bf16*)walloc((size_t)STEPS * UPD_IN * H * 2);
  __bf16* wU2  = (__bf16*)walloc((size_t)STEPS * H * H * 2);

  // 1. Pack weights to bf16 B-fragment layout (K padded to ksteps*32)
  auto pack = [&](const float* src, __bf16* dst, int srows, int ksteps, int nmat) {
    long long elems = (long long)ksteps * 8 * 32 * 16;   // = padded_K * H
    dim3 g((unsigned)((elems + 255) / 256), (unsigned)nmat);
    pack_b_bf16_kernel<<<g, 256, 0, stream>>>(src, dst, srows, ksteps);
  };
  pack(emb_W1, wE1, EMB_IN, 1, 1);
  pack(emb_W2, wE2, H,      4, 1);
  pack(msg1_W, wM1, MSG_IN, 9, STEPS);
  pack(msg2_W, wM2, H,      4, STEPS);
  pack(upd1_W, wU1, UPD_IN, 8, STEPS);
  pack(upd2_W, wU2, H,      4, STEPS);

  // 2. Zero agg + deg, compute degrees / inverse degrees
  {
    long long na = (long long)N_NODES * H;
    zero_f32_kernel<<<(unsigned)((na + 255) / 256), 256, 0, stream>>>(agg, na);
    zero_f32_kernel<<<(N_NODES + 255) / 256, 256, 0, stream>>>(deg, N_NODES);
    deg_kernel<<<(N_EDGES + 255) / 256, 256, 0, stream>>>(ei, deg, N_EDGES);
    invdeg_kernel<<<(N_NODES + 255) / 256, 256, 0, stream>>>(deg, invd, N_NODES);
  }

  // 3. Embedding
  int node_tiles = (N_NODES + 15) / 16;                  // 625
  int edge_tiles = (N_EDGES + 15) / 16;                  // 5000
  embed_kernel<<<(node_tiles + 3) / 4, 128, 0, stream>>>(
      x, pos, gattr, wE1, emb_b1, wE2, emb_b2, hf, hbf, N_NODES, node_tiles);

  // 4. Message-passing steps
  for (int s = 0; s < STEPS; ++s) {
    msg_kernel<<<(edge_tiles + 3) / 4, 128, 0, stream>>>(
        hbf, eattr, ei,
        wM1 + (size_t)s * MSG_K * H, msg1_b + (size_t)s * H,
        wM2 + (size_t)s * H * H,     msg2_b + (size_t)s * H,
        agg, N_EDGES, edge_tiles);
    upd_kernel<<<(node_tiles + 3) / 4, 128, 0, stream>>>(
        invd, agg, hf, hbf,
        wU1 + (size_t)s * UPD_IN * H, upd1_b + (size_t)s * H,
        wU2 + (size_t)s * H * H,      upd2_b + (size_t)s * H,
        N_NODES, node_tiles);
  }

  // 5. Conv head
  conv_kernel<<<(N_NODES + 3) / 4, 128, 0, stream>>>(hf, c1W, c1b, c2W, c2b, out, N_NODES);
}
